// ConvLayer_13116830122571
// MI455X (gfx1250) — matrix operand order repaired and checked by reading, not compiled, not run on hardware
//
#include <hip/hip_runtime.h>
#include <hip/hip_bf16.h>
#include <math.h>

#define F_DIM   128
#define M_NBR   12
#define K_NBR   64
#define IN_DIM  320   // 2F + K
#define OUT_DIM 256   // 2F
#define KT_COUNT 10   // 320 / 32
#define CT_COUNT 16   // 256 / 16
#define ROWS_PER_BLOCK 32  // two 16-row A tiles per block

typedef __attribute__((ext_vector_type(16))) __bf16 vbf16;
typedef __attribute__((ext_vector_type(8)))  float  vf8;

__device__ __forceinline__ __bf16 f2bf(float f) {
  unsigned u = __builtin_bit_cast(unsigned, f);
  u += 0x7FFFu + ((u >> 16) & 1u);            // round-to-nearest-even
  unsigned short s = (unsigned short)(u >> 16);
  return __builtin_bit_cast(__bf16, s);
}
__device__ __forceinline__ float bf2f(__bf16 h) {
  unsigned short s = __builtin_bit_cast(unsigned short, h);
  unsigned u = (unsigned)s << 16;
  return __builtin_bit_cast(float, u);
}

// ---------------------------------------------------------------------------
// Kernel 0: pack W [256, 320] fp32 -> bf16 B-fragments, layout [ct][kt][lane][16]
// B (32x16 bf16): lanes 0-15 hold K=0..15 of their column, lanes 16-31 hold K=16..31
// ---------------------------------------------------------------------------
__global__ void pack_w_kernel(const float* __restrict__ W, __bf16* __restrict__ wp) {
  int e = blockIdx.x * blockDim.x + threadIdx.x;
  const int TOT = CT_COUNT * KT_COUNT * 32 * 16;
  if (e >= TOT) return;
  int i    = e & 15;
  int lane = (e >> 4) & 31;
  int kt   = (e >> 9) % KT_COUNT;
  int ct   = e / (16 * 32 * KT_COUNT);
  int col  = ct * 16 + (lane & 15);
  int k    = kt * 32 + (lane >> 4) * 16 + i;
  wp[e] = f2bf(W[(size_t)col * IN_DIM + k]);
}

// ---------------------------------------------------------------------------
// Kernel 1: fused gather + concat + GEMM (bf16 WMMA) + bias -> gws bf16 [R,256]
// One block: 32 rows x 256 cols. 8 waves; each wave owns 2 column tiles and
// both 16-row A tiles, so each K-step's B registers feed 4 WMMAs (2x B reuse).
// ---------------------------------------------------------------------------
__global__ void __launch_bounds__(256) gemm_kernel(
    const float* __restrict__ atom, const float* __restrict__ nbr,
    const long long* __restrict__ idx, const __bf16* __restrict__ wp,
    const float* __restrict__ bias, __bf16* __restrict__ gws,
    int R) {
  // A staged in WMMA-fragment order: [rt][kt][lane][16 bf16] (contiguous 32B/lane)
  __shared__ __align__(32) __bf16 As[ROWS_PER_BLOCK * IN_DIM];

  const int tid  = threadIdx.x;
  const int tile = blockIdx.x;

  // ---- stage A: gather + concat + fp32->bf16, permuted into fragment layout
  for (int e = tid; e < ROWS_PER_BLOCK * IN_DIM; e += 256) {
    int row32 = e / IN_DIM;            // 0..31
    int k     = e - row32 * IN_DIM;
    long long rg = (long long)tile * ROWS_PER_BLOCK + row32;
    float v = 0.0f;
    if (rg < R) {
      int n = (int)(rg / M_NBR);
      int m = (int)(rg - (long long)n * M_NBR);
      if (k < F_DIM) {
        v = atom[(size_t)n * F_DIM + k];
      } else if (k < 2 * F_DIM) {
        long long j = idx[(size_t)n * M_NBR + m];
        v = atom[(size_t)j * F_DIM + (k - F_DIM)];
      } else {
        v = nbr[((size_t)n * M_NBR + m) * K_NBR + (k - 2 * F_DIM)];
      }
    }
    // A 16x32 bf16 layout (contiguous k, matches packed B):
    //   k5 in [0,15]  -> lanes 0-15,  elems 0-15
    //   k5 in [16,31] -> lanes 16-31, elems 0-15
    int rt   = row32 >> 4;
    int row  = row32 & 15;
    int kt   = k >> 5;
    int k5   = k & 31;
    int half = k5 >> 4;
    int elem = k5 & 15;
    int lane = half * 16 + row;
    As[(((rt * KT_COUNT) + kt) * 32 + lane) * 16 + elem] = f2bf(v);
  }
  __syncthreads();

  const int lane = tid & 31;
  const int w    = tid >> 5;
  const int ct0  = w * 2;
  const int ct1  = w * 2 + 1;

  vf8 acc00 = {};  // (rt0, ct0)
  vf8 acc01 = {};  // (rt0, ct1)
  vf8 acc10 = {};  // (rt1, ct0)
  vf8 acc11 = {};  // (rt1, ct1)
  #pragma unroll
  for (int kt = 0; kt < KT_COUNT; ++kt) {
    vbf16 b0 = *(const vbf16*)&wp[((size_t)(ct0 * KT_COUNT + kt) * 32 + lane) * 16];
    vbf16 b1 = *(const vbf16*)&wp[((size_t)(ct1 * KT_COUNT + kt) * 32 + lane) * 16];
    vbf16 a0 = *(const vbf16*)&As[((0 * KT_COUNT + kt) * 32 + lane) * 16];
    vbf16 a1 = *(const vbf16*)&As[((1 * KT_COUNT + kt) * 32 + lane) * 16];
    acc00 = __builtin_amdgcn_wmma_f32_16x16x32_bf16(false, a0, false, b0,
                                                    (short)0, acc00, false, false);
    acc01 = __builtin_amdgcn_wmma_f32_16x16x32_bf16(false, a0, false, b1,
                                                    (short)0, acc01, false, false);
    acc10 = __builtin_amdgcn_wmma_f32_16x16x32_bf16(false, a1, false, b0,
                                                    (short)0, acc10, false, false);
    acc11 = __builtin_amdgcn_wmma_f32_16x16x32_bf16(false, a1, false, b1,
                                                    (short)0, acc11, false, false);
  }

  // D layout: lane holds column (lane&15); VGPR v holds row v + 8*(lane>>4)
  const int colL = lane & 15;
  const int rh   = lane >> 4;
  const float bias0 = bias[ct0 * 16 + colL];
  const float bias1 = bias[ct1 * 16 + colL];
  #pragma unroll
  for (int v = 0; v < 8; ++v) {
    long long rg0 = (long long)tile * ROWS_PER_BLOCK + v + 8 * rh;        // rt=0
    long long rg1 = rg0 + 16;                                             // rt=1
    if (rg0 < R) {
      size_t base = (size_t)rg0 * OUT_DIM;
      gws[base + ct0 * 16 + colL] = f2bf(acc00[v] + bias0);
      gws[base + ct1 * 16 + colL] = f2bf(acc01[v] + bias1);
    }
    if (rg1 < R) {
      size_t base = (size_t)rg1 * OUT_DIM;
      gws[base + ct0 * 16 + colL] = f2bf(acc10[v] + bias0);
      gws[base + ct1 * 16 + colL] = f2bf(acc11[v] + bias1);
    }
  }
}

// ---------------------------------------------------------------------------
// Kernel 2: BN1 per-channel partial sums/sumsq over R rows (deterministic)
// ---------------------------------------------------------------------------
__global__ void __launch_bounds__(256) stats1_kernel(const __bf16* __restrict__ gws,
                                                     float* __restrict__ part, int R) {
  int c = threadIdx.x;
  float s = 0.0f, sq = 0.0f;
  for (long long r = blockIdx.x; r < R; r += gridDim.x) {
    float v = bf2f(gws[(size_t)r * OUT_DIM + c]);
    s += v; sq += v * v;
  }
  part[(size_t)blockIdx.x * 512 + c]       = s;
  part[(size_t)blockIdx.x * 512 + 256 + c] = sq;
}

__global__ void reduce1_kernel(const float* __restrict__ part,
                               const float* __restrict__ gamma,
                               const float* __restrict__ beta,
                               float* __restrict__ params,
                               int nblocks, long long count) {
  int c = threadIdx.x;                       // 0..255
  float s = 0.0f, sq = 0.0f;
  for (int b = 0; b < nblocks; ++b) {
    s  += part[(size_t)b * 512 + c];
    sq += part[(size_t)b * 512 + 256 + c];
  }
  float mean  = s / (float)count;
  float var   = sq / (float)count - mean * mean;
  float scale = gamma[c] * rsqrtf(var + 1e-5f);
  params[c]        = scale;
  params[256 + c]  = beta[c] - mean * scale;
}

// ---------------------------------------------------------------------------
// Kernel 3: apply BN1, sigmoid(filter)*relu(core), sum over M -> sums [N,128]
// ---------------------------------------------------------------------------
__global__ void __launch_bounds__(128) gate_sum_kernel(const __bf16* __restrict__ gws,
                                                       const float* __restrict__ params,
                                                       float* __restrict__ sums, int N) {
  int n = blockIdx.x;
  int f = threadIdx.x;
  if (n >= N) return;
  float sF = params[f],        hF = params[256 + f];
  float sC = params[128 + f],  hC = params[384 + f];
  float acc = 0.0f;
  size_t base = (size_t)n * M_NBR * OUT_DIM;
  #pragma unroll
  for (int m = 0; m < M_NBR; ++m) {
    float gf = bf2f(gws[base + m * OUT_DIM + f])       * sF + hF;
    float gc = bf2f(gws[base + m * OUT_DIM + 128 + f]) * sC + hC;
    float sig = 1.0f / (1.0f + __expf(-gf));
    acc += sig * fmaxf(gc, 0.0f);
  }
  sums[(size_t)n * F_DIM + f] = acc;
}

// ---------------------------------------------------------------------------
// Kernel 4/5: BN2 stats + params
// ---------------------------------------------------------------------------
__global__ void __launch_bounds__(128) stats2_kernel(const float* __restrict__ sums,
                                                     float* __restrict__ part, int N) {
  int c = threadIdx.x;
  float s = 0.0f, sq = 0.0f;
  for (long long n = blockIdx.x; n < N; n += gridDim.x) {
    float v = sums[(size_t)n * F_DIM + c];
    s += v; sq += v * v;
  }
  part[(size_t)blockIdx.x * 256 + c]       = s;
  part[(size_t)blockIdx.x * 256 + 128 + c] = sq;
}

__global__ void reduce2_kernel(const float* __restrict__ part,
                               const float* __restrict__ gamma,
                               const float* __restrict__ beta,
                               float* __restrict__ params,
                               int nblocks, long long count) {
  int c = threadIdx.x;                       // 0..127
  float s = 0.0f, sq = 0.0f;
  for (int b = 0; b < nblocks; ++b) {
    s  += part[(size_t)b * 256 + c];
    sq += part[(size_t)b * 256 + 128 + c];
  }
  float mean  = s / (float)count;
  float var   = sq / (float)count - mean * mean;
  float scale = gamma[c] * rsqrtf(var + 1e-5f);
  params[c]        = scale;
  params[128 + c]  = beta[c] - mean * scale;
}

// ---------------------------------------------------------------------------
// Kernel 6: out = relu(atom + BN2(sums))
// ---------------------------------------------------------------------------
__global__ void final_kernel(const float* __restrict__ atom,
                             const float* __restrict__ sums,
                             const float* __restrict__ params2,
                             float* __restrict__ out, long long total) {
  long long i = (long long)blockIdx.x * blockDim.x + threadIdx.x;
  if (i >= total) return;
  int f = (int)(i & (F_DIM - 1));
  float v = atom[i] + (sums[i] * params2[f] + params2[128 + f]);
  out[i] = fmaxf(v, 0.0f);
}

// ---------------------------------------------------------------------------
extern "C" void kernel_launch(void* const* d_in, const int* in_sizes, int n_in,
                              void* d_out, int out_size, void* d_ws, size_t ws_size,
                              hipStream_t stream) {
  const float*     atom = (const float*)d_in[0];
  const float*     nbr  = (const float*)d_in[1];
  const long long* idx  = (const long long*)d_in[2];
  const float*     W    = (const float*)d_in[3];
  const float*     bias = (const float*)d_in[4];
  const float*     g1   = (const float*)d_in[5];
  const float*     b1   = (const float*)d_in[6];
  const float*     g2   = (const float*)d_in[7];
  const float*     b2   = (const float*)d_in[8];

  const int N = in_sizes[0] / F_DIM;
  const long long R = (long long)N * M_NBR;

  // workspace carve-up (256B aligned)
  char* ws = (char*)d_ws;
  size_t off = 0;
  auto carve = [&](size_t bytes) {
    char* p = ws + off;
    off = (off + bytes + 255) & ~(size_t)255;
    return p;
  };
  __bf16* wp      = (__bf16*)carve((size_t)CT_COUNT * KT_COUNT * 32 * 16 * 2);
  __bf16* gws     = (__bf16*)carve((size_t)R * OUT_DIM * 2);
  float*  sums    = (float*) carve((size_t)N * F_DIM * 4);
  float*  part1   = (float*) carve((size_t)1024 * 512 * 4);
  float*  part2   = (float*) carve((size_t)1024 * 256 * 4);
  float*  params1 = (float*) carve(512 * 4);
  float*  params2 = (float*) carve(256 * 4);
  (void)ws_size; (void)n_in; (void)out_size;

  const int packTot = CT_COUNT * KT_COUNT * 32 * 16;
  pack_w_kernel<<<(packTot + 255) / 256, 256, 0, stream>>>(W, wp);

  const int tiles = (int)((R + ROWS_PER_BLOCK - 1) / ROWS_PER_BLOCK);
  gemm_kernel<<<tiles, 256, 0, stream>>>(atom, nbr, idx, wp, bias, gws, (int)R);

  stats1_kernel<<<1024, 256, 0, stream>>>(gws, part1, (int)R);
  reduce1_kernel<<<1, 256, 0, stream>>>(part1, g1, b1, params1, 1024, R);

  gate_sum_kernel<<<N, 128, 0, stream>>>(gws, params1, sums, N);

  stats2_kernel<<<1024, 128, 0, stream>>>(sums, part2, N);
  reduce2_kernel<<<1, 128, 0, stream>>>(part2, g2, b2, params2, 1024, (long long)N);

  const long long total = (long long)N * F_DIM;
  final_kernel<<<(int)((total + 255) / 256), 256, 0, stream>>>(atom, sums, params2,
                                                               (float*)d_out, total);
}
